// NT_Xent_14456859918371
// MI455X (gfx1250) — matrix operand order repaired and checked by reading, not compiled, not run on hardware
//
#include <hip/hip_runtime.h>
#include <hip/hip_bf16.h>

typedef _Float16 v4h  __attribute__((ext_vector_type(4)));
typedef _Float16 v8h  __attribute__((ext_vector_type(8)));
typedef _Float16 v16h __attribute__((ext_vector_type(16)));
typedef float    v8f  __attribute__((ext_vector_type(8)));

#define B_TOTAL     8192
#define N_FEAT      128
#define LDS_STRIDE  136        // 128 + 8 halves pad -> 272B row stride (breaks bank periodicity)
#define INV_TAU     2.0f
#define LOG2E       1.4426950408889634f
#define LN2         0.6931471805599453f
#define EXP_SCALE   (INV_TAU * LOG2E)   // baked into the A operand

// ---------------- kernel 1: f32 -> f16 convert (plain + pre-scaled copies) ----------------
__global__ __launch_bounds__(256) void convert_f16(const float* __restrict__ z,
                                                   _Float16* __restrict__ zh,
                                                   _Float16* __restrict__ zhs) {
    int i = (blockIdx.x * 256 + threadIdx.x) * 4;
    float4 v = *(const float4*)(z + i);
    v4h h  = { (_Float16)v.x, (_Float16)v.y, (_Float16)v.z, (_Float16)v.w };
    v4h hs = { (_Float16)(v.x * EXP_SCALE), (_Float16)(v.y * EXP_SCALE),
               (_Float16)(v.z * EXP_SCALE), (_Float16)(v.w * EXP_SCALE) };
    *(v4h*)(zh  + i) = h;
    *(v4h*)(zhs + i) = hs;
}

// B-fragment load: 32x16 f16 tile, lane 0-15: N=lane,K=0..7,16..23; lane 16-31: N,K=8..15,24..31
__device__ __forceinline__ v16h load_bfrag(const _Float16* bp, int kk, int hi) {
    v8h b0 = *(const v8h*)(bp + kk * 32 + hi * 8);
    v8h b1 = *(const v8h*)(bp + kk * 32 + 16 + hi * 8);
    return __builtin_shufflevector(b0, b1, 0,1,2,3,4,5,6,7,8,9,10,11,12,13,14,15);
}

// ---------------- kernel 2: fused WMMA sim + exp row-sum ----------------
// grid = (32 row-blocks of 256 rows, 8 col-groups of 1024 cols), 128 threads = 4 wave32.
// Each wave owns an M=64 register tile (4 A-frag sets, 128 VGPRs): each streamed B
// fragment feeds 4 independent WMMAs -> 0.25 KB LDS traffic per WMMA.
// B fragments are rotated through a 2-deep buffer so LDS latency hides under the
// previous 4-WMMA group. A operand pre-scaled by INV_TAU*LOG2E -> acc is the exp2 arg.
__global__ __launch_bounds__(128)
void ntxent_sim_rowsum(const _Float16* __restrict__ zh,
                       const _Float16* __restrict__ zhs,
                       float* __restrict__ partial) {
    __shared__ __align__(16) _Float16 Zr[256 * LDS_STRIDE];  // scaled rows (A)
    __shared__ __align__(16) _Float16 Zc[128 * LDS_STRIDE];  // plain cols (B)

    const int tid  = threadIdx.x;        // 0..127
    const int wave = tid >> 5;           // 0..3
    const int lane = tid & 31;
    const int lo16 = lane & 15;
    const int hi   = lane >> 4;          // selects K-half per ISA fragment layout

    const int r0 = blockIdx.x * 256;     // row tile base
    const int c0 = blockIdx.y * 1024;    // col group base

    // ---- stage scaled row tile (256 x 128 f16) into LDS ----
    {
        const uint4* src = (const uint4*)(zhs + (size_t)r0 * N_FEAT);
        #pragma unroll
        for (int i = 0; i < 32; ++i) {
            int chunk = tid + 128 * i;              // 4096 chunks of 16B
            int row = chunk >> 4, c8 = chunk & 15;
            *(uint4*)&Zr[row * LDS_STRIDE + c8 * 8] = src[chunk];
        }
    }
    __syncthreads();

    // ---- A fragments for 4 strips (rows wave*64 .. wave*64+63), fixed for whole kernel ----
    v16h afrag[4][4];
    #pragma unroll
    for (int s = 0; s < 4; ++s) {
        const _Float16* ap = &Zr[(wave * 64 + s * 16 + lo16) * LDS_STRIDE];
        #pragma unroll
        for (int kk = 0; kk < 4; ++kk) {
            v8h lo = *(const v8h*)(ap + kk * 32 + hi * 8);
            v8h hh = *(const v8h*)(ap + kk * 32 + 16 + hi * 8);
            afrag[s][kk] = __builtin_shufflevector(lo, hh,
                0,1,2,3,4,5,6,7,8,9,10,11,12,13,14,15);
        }
    }

    float rowsum[4][8];
    #pragma unroll
    for (int s = 0; s < 4; ++s)
        #pragma unroll
        for (int v = 0; v < 8; ++v) rowsum[s][v] = 0.0f;

    for (int ct = 0; ct < 8; ++ct) {
        __syncthreads();   // previous Zc consumers done
        const uint4* src = (const uint4*)(zh + (size_t)(c0 + ct * 128) * N_FEAT);
        #pragma unroll
        for (int i = 0; i < 16; ++i) {
            int chunk = tid + 128 * i;              // 2048 chunks of 16B
            int row = chunk >> 4, c8 = chunk & 15;
            *(uint4*)&Zc[row * LDS_STRIDE + c8 * 8] = src[chunk];
        }
        if (ct < 7)  // CDNA5 global_prefetch_b8 for the next column tile (256B per thread)
            __builtin_prefetch(zh + (size_t)(c0 + (ct + 1) * 128) * N_FEAT + tid * 128, 0, 1);
        __syncthreads();

        #pragma unroll
        for (int ns = 0; ns < 8; ++ns) {
            const _Float16* bp = &Zc[(ns * 16 + lo16) * LDS_STRIDE];
            v8f acc[4];
            #pragma unroll
            for (int s = 0; s < 4; ++s) acc[s] = (v8f){0.f,0.f,0.f,0.f,0.f,0.f,0.f,0.f};

            // software-pipelined B stream: issue kk+1 loads before consuming kk
            v16h bcur = load_bfrag(bp, 0, hi);
            #pragma unroll
            for (int kk = 0; kk < 4; ++kk) {
                v16h bnext;
                if (kk < 3) bnext = load_bfrag(bp, kk + 1, hi);
                #pragma unroll
                for (int s = 0; s < 4; ++s)   // 4 independent XDL chains per B fragment
                    acc[s] = __builtin_amdgcn_wmma_f32_16x16x32_f16(
                               false, afrag[s][kk], false, bcur, (short)0, acc[s], false, false);
                if (kk < 3) bcur = bnext;
            }

            // batch all exps first (independent TRANS stream), then the adds: no dead slots
            float e[4][8];
            #pragma unroll
            for (int s = 0; s < 4; ++s)
                #pragma unroll
                for (int v = 0; v < 8; ++v) e[s][v] = __builtin_amdgcn_exp2f(acc[s][v]);
            #pragma unroll
            for (int s = 0; s < 4; ++s)
                #pragma unroll
                for (int v = 0; v < 8; ++v) rowsum[s][v] += e[s][v];
        }
    }

    // reduce across the 16 lanes of each half (columns), store per-row partials
    #pragma unroll
    for (int s = 0; s < 4; ++s) {
        #pragma unroll
        for (int v = 0; v < 8; ++v) {
            float sum = rowsum[s][v];
            sum += __shfl_xor(sum, 1, 32);
            sum += __shfl_xor(sum, 2, 32);
            sum += __shfl_xor(sum, 4, 32);
            sum += __shfl_xor(sum, 8, 32);
            if (lo16 == 0) {
                int rglob = r0 + wave * 64 + s * 16 + v + hi * 8;
                partial[blockIdx.y * B_TOTAL + rglob] = sum;
            }
        }
    }
}

// ---------------- kernel 3: per-row contribution (deterministic) ----------------
__global__ __launch_bounds__(256)
void ntxent_finalize(const float* __restrict__ z,
                     const _Float16* __restrict__ zh,
                     const _Float16* __restrict__ zhs,
                     const float* __restrict__ partial,
                     float* __restrict__ contrib) {
    int b = blockIdx.x * 256 + threadIdx.x;      // 0..8191
    float neg = 0.0f;
    #pragma unroll
    for (int g = 0; g < 8; ++g) neg += partial[g * B_TOTAL + b];

    // subtract diagonal term exp2(zhs[b] . zh[b]) -- same f16 values the WMMA consumed
    {
        const v8h* hs = (const v8h*)(zhs + (size_t)b * N_FEAT);
        const v8h* hu = (const v8h*)(zh  + (size_t)b * N_FEAT);
        float sd = 0.0f;
        #pragma unroll
        for (int c = 0; c < N_FEAT / 8; ++c) {
            v8h a = hs[c], u = hu[c];
            #pragma unroll
            for (int e = 0; e < 8; ++e) sd = fmaf((float)a[e], (float)u[e], sd);
        }
        neg -= __builtin_amdgcn_exp2f(sd);
    }

    int p = b >> 1;  // positive pair rows (2p, 2p+1), dot in full f32
    const float* zi = z + (size_t)(2 * p) * N_FEAT;
    const float* zj = z + (size_t)(2 * p + 1) * N_FEAT;
    float dot = 0.0f;
    #pragma unroll 4
    for (int n = 0; n < N_FEAT; n += 4) {
        float4 a = *(const float4*)(zi + n);
        float4 c = *(const float4*)(zj + n);
        dot = fmaf(a.x, c.x, dot);
        dot = fmaf(a.y, c.y, dot);
        dot = fmaf(a.z, c.z, dot);
        dot = fmaf(a.w, c.w, dot);
    }
    // contribution = log(neg) - log(pos);  log(pos) = dot/tau
    contrib[b] = __builtin_amdgcn_logf(neg) * LN2 - dot * INV_TAU;
}

// ---------------- kernel 4: deterministic tree reduction -> scalar loss ----------------
__global__ __launch_bounds__(1024)
void ntxent_reduce(const float* __restrict__ contrib, float* __restrict__ out) {
    __shared__ float s[1024];
    int t = threadIdx.x;
    float v = 0.0f;
    #pragma unroll
    for (int i = 0; i < 8; ++i) v += contrib[t + 1024 * i];
    s[t] = v;
    __syncthreads();
    for (int off = 512; off > 0; off >>= 1) {
        if (t < off) s[t] += s[t + off];
        __syncthreads();
    }
    if (t == 0) out[0] = s[0] * (1.0f / (float)B_TOTAL);
}

extern "C" void kernel_launch(void* const* d_in, const int* in_sizes, int n_in,
                              void* d_out, int out_size, void* d_ws, size_t ws_size,
                              hipStream_t stream) {
    const float* z = (const float*)d_in[0];
    float* out = (float*)d_out;

    // workspace layout
    const size_t ZH_BYTES = (size_t)B_TOTAL * N_FEAT * sizeof(_Float16);   // 2 MB each
    _Float16* zh      = (_Float16*)d_ws;
    _Float16* zhs     = (_Float16*)((char*)d_ws + ZH_BYTES);
    float*    partial = (float*)((char*)d_ws + 2 * ZH_BYTES);              // 8*8192 f32
    float*    contrib = partial + 8 * B_TOTAL;                             // 8192 f32

    convert_f16<<<(B_TOTAL * N_FEAT) / (256 * 4), 256, 0, stream>>>(z, zh, zhs);

    dim3 grid(32, 8);
    ntxent_sim_rowsum<<<grid, 128, 0, stream>>>(zh, zhs, partial);

    ntxent_finalize<<<B_TOTAL / 256, 256, 0, stream>>>(z, zh, zhs, partial, contrib);
    ntxent_reduce<<<1, 1024, 0, stream>>>(contrib, out);
}
